// GRU_88708254532000
// MI455X (gfx1250) — matrix operand order repaired and checked by reading, not compiled, not run on hardware
//
#include <hip/hip_runtime.h>

// ---------------------------------------------------------------------------
// GRU (2-layer) + logits on gfx1250 using v_wmma_f32_16x16x32_bf16.
// Weights are pre-swizzled once per call into WMMA B-fragment order:
//   fragment(ktile, ntile) = 32 lanes x 16 bf16 (1KB), lane-contiguous,
//   element e of lane L = B[k = ktile*32 + (e/8)*16 + (L/16)*8 + e%8][n = ntile*16 + L%16]
// A-fragments follow the ISA 16-bit A 16x32 layout: lane L covers row M=L%16,
// elements 0..7 -> K = k0 + (L/16)*8 + e, elements 8..15 -> K = k0+16+(L/16)*8.
// B-fragment loads are uniform-base + constant-offset + lane*32B, so they
// lower to saddr-based global_load_b128 with immediate offsets (no per-
// iteration address VALU); loads are batched ahead of the WMMAs so the
// scheduler can keep several b128 loads in flight per matrix op.
// ---------------------------------------------------------------------------

typedef __attribute__((ext_vector_type(16))) __bf16 v16bf;
typedef __attribute__((ext_vector_type(8)))  float  v8f;
typedef __attribute__((ext_vector_type(4)))  unsigned int u32x4;

union AFrag { v16bf v; u32x4 q[2]; };

static __device__ __forceinline__ unsigned short f2bf(float f) {
  union { float f; unsigned int u; } c; c.f = f;
  unsigned int u = c.u + 0x7FFFu + ((c.u >> 16) & 1u);   // round-to-nearest-even
  return (unsigned short)(u >> 16);
}

static __device__ __forceinline__ float sigmoidf_(float x) {
  return 1.0f / (1.0f + __expf(-x));
}

static __device__ __forceinline__ v8f wmma_bf16(v16bf a, v16bf b, v8f c) {
  return __builtin_amdgcn_wmma_f32_16x16x32_bf16(false, a, false, b, (short)0, c,
                                                 false, false);
}

// --------------------------- prep kernels ----------------------------------

// fp32 row-major [K x N] weight  ->  bf16 WMMA B-fragment order (N padded).
__global__ void swizzle_w(const float* __restrict__ src, unsigned short* __restrict__ dst,
                          int K, int N, long long total) {
  long long id = (long long)blockIdx.x * blockDim.x + threadIdx.x;
  if (id >= total) return;
  int e    = (int)(id & 15);
  int lane = (int)((id >> 4) & 31);
  long long frag = id >> 9;
  long long KT = (long long)(K >> 5);
  int kt = (int)(frag % KT);
  int nt = (int)(frag / KT);
  int k = (kt << 5) + ((e >> 3) << 4) + ((lane >> 4) << 3) + (e & 7);
  int n = (nt << 4) + (lane & 15);
  float val = (n < N) ? src[(size_t)k * N + n] : 0.0f;
  dst[id] = f2bf(val);
}

// Gather all timestep embeddings up-front: e_bf16[t][b][k] = emb[inputs[t][b]][k]
__global__ void embed_gather(const int* __restrict__ inputs, const float* __restrict__ emb,
                             unsigned short* __restrict__ out, long long total) {
  long long id = (long long)blockIdx.x * blockDim.x + threadIdx.x;
  if (id >= total) return;
  int k = (int)(id & 511);          // E = 512
  long long be = id >> 9;           // t*B + b
  int tok = inputs[be];
  out[id] = f2bf(emb[(size_t)tok * 512 + k]);
}

__global__ void init_h(const float* __restrict__ hidden, float* __restrict__ hf,
                       unsigned short* __restrict__ hb, int total) {
  int id = blockIdx.x * blockDim.x + threadIdx.x;
  if (id >= total) return;
  float v = hidden[id];
  hf[id] = v;
  hb[id] = f2bf(v);
}

__global__ void copy_h(const float* __restrict__ hf, float* __restrict__ out, int total) {
  int id = blockIdx.x * blockDim.x + threadIdx.x;
  if (id < total) out[id] = hf[id];
}

// --------------------------- per-step kernels ------------------------------
// Grid: 16 blocks (64-col tiles over H=1024) x 128 threads (4 waves x 16 rows).

// r = sigmoid(Xa@Wr + bWr + Hprev@Ur + bUr); z likewise; gx2 = Xa@Wh + bWh.
// Outputs: z (fp32), gx2 (fp32), rh = bf16(r * Hprev_f32).
template <int KTIN>   // K-tiles of the input projection: Kin = KTIN*32
__global__ __launch_bounds__(128) void gru_gates(
    const unsigned short* __restrict__ Xa,
    const unsigned short* __restrict__ Hb, const float* __restrict__ Hf,
    const unsigned short* __restrict__ Wr, const unsigned short* __restrict__ Wz,
    const unsigned short* __restrict__ Wh,
    const unsigned short* __restrict__ Ur, const unsigned short* __restrict__ Uz,
    const float* __restrict__ bWr, const float* __restrict__ bUr,
    const float* __restrict__ bWz, const float* __restrict__ bUz,
    const float* __restrict__ bWh,
    float* __restrict__ z_out, float* __restrict__ gx2_out,
    unsigned short* __restrict__ rh_out) {
  const int lane = threadIdx.x & 31;
  const int half = lane >> 4;
  const int mrow = lane & 15;
  const int m0   = (threadIdx.x >> 5) << 4;
  const int ntB  = blockIdx.x << 2;
  const size_t laneB = (size_t)lane * 16;

  // uniform bases (SGPR-friendly) + per-lane term; frag offsets become imms
  const unsigned short* WrB = Wr + (size_t)ntB * KTIN * 512 + laneB;
  const unsigned short* WzB = Wz + (size_t)ntB * KTIN * 512 + laneB;
  const unsigned short* WhB = Wh + (size_t)ntB * KTIN * 512 + laneB;
  const unsigned short* UrB = Ur + (size_t)ntB * 32 * 512 + laneB;
  const unsigned short* UzB = Uz + (size_t)ntB * 32 * 512 + laneB;
  const unsigned short* aX  = Xa + (size_t)(m0 + mrow) * (KTIN * 32) + (half << 3);
  const unsigned short* aH  = Hb + (size_t)(m0 + mrow) * 1024 + (half << 3);

  v8f accR[4] = {}; v8f accZ[4] = {}; v8f accH[4] = {};

#pragma unroll 4
  for (int kt = 0; kt < KTIN; ++kt) {            // input projections
    AFrag a;
    a.q[0] = *(const u32x4*)(aX + kt * 32);
    a.q[1] = *(const u32x4*)(aX + kt * 32 + 16);
#pragma unroll
    for (int nf = 0; nf < 4; ++nf) {
      const size_t fo = ((size_t)nf * KTIN + kt) * 512;
      v16bf bR = *(const v16bf*)(WrB + fo);
      v16bf bZ = *(const v16bf*)(WzB + fo);
      v16bf bH = *(const v16bf*)(WhB + fo);
      accR[nf] = wmma_bf16(a.v, bR, accR[nf]);
      accZ[nf] = wmma_bf16(a.v, bZ, accZ[nf]);
      accH[nf] = wmma_bf16(a.v, bH, accH[nf]);
    }
  }
#pragma unroll 4
  for (int kt = 0; kt < 32; ++kt) {              // recurrent projections (K = 1024)
    AFrag a;
    a.q[0] = *(const u32x4*)(aH + kt * 32);
    a.q[1] = *(const u32x4*)(aH + kt * 32 + 16);
#pragma unroll
    for (int nf = 0; nf < 4; ++nf) {
      const size_t fo = ((size_t)nf * 32 + kt) * 512;
      v16bf bR = *(const v16bf*)(UrB + fo);
      v16bf bZ = *(const v16bf*)(UzB + fo);
      accR[nf] = wmma_bf16(a.v, bR, accR[nf]);
      accZ[nf] = wmma_bf16(a.v, bZ, accZ[nf]);
    }
  }
#pragma unroll
  for (int nf = 0; nf < 4; ++nf) {
    const int n  = ((ntB + nf) << 4) + mrow;
    const float br = bWr[n] + bUr[n];
    const float bz = bWz[n] + bUz[n];
    const float bh = bWh[n];
#pragma unroll
    for (int v = 0; v < 8; ++v) {
      const int m = m0 + (half << 3) + v;        // C/D layout: VGPR v, half L/16
      const size_t idx = (size_t)m * 1024 + n;
      const float r  = sigmoidf_(accR[nf][v] + br);
      const float zz = sigmoidf_(accZ[nf][v] + bz);
      z_out[idx]   = zz;
      gx2_out[idx] = accH[nf][v] + bh;
      rh_out[idx]  = f2bf(r * Hf[idx]);
    }
  }
}

// hm = tanh(gx2 + rh@Uh + bUh);  combine:
//   deeper==0 (layer 0): hnew = (1-z)*hm + z*hp   (reference asymmetry)
//   deeper==1          : hnew =   z  *hm + (1-z)*hp
__global__ __launch_bounds__(128) void gru_cand(
    const unsigned short* __restrict__ rh, const unsigned short* __restrict__ Uh,
    const float* __restrict__ bUh,
    const float* __restrict__ gx2, const float* __restrict__ zv,
    const float* __restrict__ Hf_prev,
    float* __restrict__ Hf_new, unsigned short* __restrict__ Hb_new, int deeper) {
  const int lane = threadIdx.x & 31;
  const int half = lane >> 4;
  const int mrow = lane & 15;
  const int m0   = (threadIdx.x >> 5) << 4;
  const int ntB  = blockIdx.x << 2;
  const size_t laneB = (size_t)lane * 16;

  const unsigned short* UhB = Uh + (size_t)ntB * 32 * 512 + laneB;
  const unsigned short* aP  = rh + (size_t)(m0 + mrow) * 1024 + (half << 3);

  v8f acc[4] = {};
#pragma unroll 4
  for (int kt = 0; kt < 32; ++kt) {
    AFrag a;
    a.q[0] = *(const u32x4*)(aP + kt * 32);
    a.q[1] = *(const u32x4*)(aP + kt * 32 + 16);
    v16bf b0 = *(const v16bf*)(UhB + ((size_t)0 * 32 + kt) * 512);
    v16bf b1 = *(const v16bf*)(UhB + ((size_t)1 * 32 + kt) * 512);
    v16bf b2 = *(const v16bf*)(UhB + ((size_t)2 * 32 + kt) * 512);
    v16bf b3 = *(const v16bf*)(UhB + ((size_t)3 * 32 + kt) * 512);
    acc[0] = wmma_bf16(a.v, b0, acc[0]);
    acc[1] = wmma_bf16(a.v, b1, acc[1]);
    acc[2] = wmma_bf16(a.v, b2, acc[2]);
    acc[3] = wmma_bf16(a.v, b3, acc[3]);
  }
#pragma unroll
  for (int nf = 0; nf < 4; ++nf) {
    const int n = ((ntB + nf) << 4) + mrow;
    const float bh = bUh[n];
#pragma unroll
    for (int v = 0; v < 8; ++v) {
      const int m = m0 + (half << 3) + v;
      const size_t idx = (size_t)m * 1024 + n;
      const float hm = tanhf(acc[nf][v] + bh + gx2[idx]);
      const float z  = zv[idx];
      const float hp = Hf_prev[idx];
      const float hn = deeper ? (z * hm + (1.0f - z) * hp)
                              : ((1.0f - z) * hm + z * hp);
      Hf_new[idx] = hn;
      Hb_new[idx] = f2bf(hn);
    }
  }
}

// logits_t[64 x 10000] = h1 @ Wy + by.  Grid: 157 blocks (64-col tiles over 10048 pad).
__global__ __launch_bounds__(128) void gru_logits(
    const unsigned short* __restrict__ h1, const unsigned short* __restrict__ Wys,
    const float* __restrict__ by, float* __restrict__ out) {
  const int lane = threadIdx.x & 31;
  const int half = lane >> 4;
  const int mrow = lane & 15;
  const int m0   = (threadIdx.x >> 5) << 4;
  const int ntB  = blockIdx.x << 2;
  const size_t laneB = (size_t)lane * 16;

  const unsigned short* WyB = Wys + (size_t)ntB * 32 * 512 + laneB;
  const unsigned short* aP  = h1 + (size_t)(m0 + mrow) * 1024 + (half << 3);

  v8f acc[4] = {};
#pragma unroll 4
  for (int kt = 0; kt < 32; ++kt) {
    AFrag a;
    a.q[0] = *(const u32x4*)(aP + kt * 32);
    a.q[1] = *(const u32x4*)(aP + kt * 32 + 16);
    v16bf b0 = *(const v16bf*)(WyB + ((size_t)0 * 32 + kt) * 512);
    v16bf b1 = *(const v16bf*)(WyB + ((size_t)1 * 32 + kt) * 512);
    v16bf b2 = *(const v16bf*)(WyB + ((size_t)2 * 32 + kt) * 512);
    v16bf b3 = *(const v16bf*)(WyB + ((size_t)3 * 32 + kt) * 512);
    acc[0] = wmma_bf16(a.v, b0, acc[0]);
    acc[1] = wmma_bf16(a.v, b1, acc[1]);
    acc[2] = wmma_bf16(a.v, b2, acc[2]);
    acc[3] = wmma_bf16(a.v, b3, acc[3]);
  }
#pragma unroll
  for (int nf = 0; nf < 4; ++nf) {
    const int n = ((ntB + nf) << 4) + mrow;
    if (n < 10000) {
      const float b = by[n];
#pragma unroll
      for (int v = 0; v < 8; ++v) {
        const int m = m0 + (half << 3) + v;
        out[(size_t)m * 10000 + n] = acc[nf][v] + b;
      }
    }
  }
}

// ------------------------------- host --------------------------------------

extern "C" void kernel_launch(void* const* d_in, const int* in_sizes, int n_in,
                              void* d_out, int out_size, void* d_ws, size_t ws_size,
                              hipStream_t stream) {
  (void)in_sizes; (void)n_in; (void)out_size; (void)ws_size;
  const int*   inputs = (const int*)d_in[0];
  const float* hidden = (const float*)d_in[1];
  const float* emb    = (const float*)d_in[2];
  const float* W0     = (const float*)d_in[3];
  const float* U0     = (const float*)d_in[4];
  const float* bW0    = (const float*)d_in[5];
  const float* bU0    = (const float*)d_in[6];
  const float* Wl     = (const float*)d_in[7];
  const float* Ul     = (const float*)d_in[8];
  const float* bWl    = (const float*)d_in[9];
  const float* bUl    = (const float*)d_in[10];
  const float* Wy     = (const float*)d_in[11];
  const float* by     = (const float*)d_in[12];
  float* out = (float*)d_out;

  // ---- workspace carve-out (256B aligned) ----
  char* ws = (char*)d_ws;
  size_t off = 0;
  auto alloc = [&](size_t bytes) -> void* {
    void* p = ws + off;
    off = (off + bytes + 255) & ~(size_t)255;
    return p;
  };
  const size_t sW0 = (size_t)16 * 64 * 512;   // frags per W0 gate  (K=512,  N=1024)
  const size_t sU  = (size_t)32 * 64 * 512;   // frags per U/Wl/Ul gate (K=1024)
  unsigned short* eB  = (unsigned short*)alloc((size_t)128 * 64 * 512 * 2);
  unsigned short* W0s = (unsigned short*)alloc(3 * sW0 * 2);
  unsigned short* U0s = (unsigned short*)alloc(3 * sU * 2);
  unsigned short* Wls = (unsigned short*)alloc(3 * sU * 2);
  unsigned short* Uls = (unsigned short*)alloc(3 * sU * 2);
  unsigned short* Wys = (unsigned short*)alloc((size_t)32 * 628 * 512 * 2); // V pad 10048
  float*          hf  = (float*)alloc((size_t)2 * 64 * 1024 * 4);
  unsigned short* hb  = (unsigned short*)alloc((size_t)2 * 64 * 1024 * 2);
  float*          zws = (float*)alloc((size_t)64 * 1024 * 4);
  float*          gx2 = (float*)alloc((size_t)64 * 1024 * 4);
  unsigned short* rh  = (unsigned short*)alloc((size_t)64 * 1024 * 2);

  // ---- one-time prep (re-run each call; deterministic) ----
  auto swz = [&](const float* src, unsigned short* dst, int K, int N, int NT) {
    long long total = (long long)(K >> 5) * NT * 512;
    int blocks = (int)((total + 255) / 256);
    swizzle_w<<<blocks, 256, 0, stream>>>(src, dst, K, N, total);
  };
  for (int g = 0; g < 3; ++g) {
    swz(W0 + (size_t)g * 512 * 1024,  W0s + (size_t)g * sW0, 512,  1024, 64);
    swz(U0 + (size_t)g * 1024 * 1024, U0s + (size_t)g * sU,  1024, 1024, 64);
    swz(Wl + (size_t)g * 1024 * 1024, Wls + (size_t)g * sU,  1024, 1024, 64);
    swz(Ul + (size_t)g * 1024 * 1024, Uls + (size_t)g * sU,  1024, 1024, 64);
  }
  swz(Wy, Wys, 1024, 10000, 628);

  {
    long long total = (long long)128 * 64 * 512;
    embed_gather<<<(int)((total + 255) / 256), 256, 0, stream>>>(inputs, emb, eB, total);
  }
  init_h<<<512, 256, 0, stream>>>(hidden, hf, hb, 2 * 64 * 1024);

  float*          h0f = hf;
  float*          h1f = hf + (size_t)64 * 1024;
  unsigned short* h0b = hb;
  unsigned short* h1b = hb + (size_t)64 * 1024;

  // ---- sequential scan over 128 timesteps ----
  for (int t = 0; t < 128; ++t) {
    const unsigned short* et = eB + (size_t)t * 64 * 512;
    // layer 0 (gates use W0[0..2], U0[0..1]; candidate uses U0[2])
    gru_gates<16><<<16, 128, 0, stream>>>(
        et, h0b, h0f,
        W0s + 0 * sW0, W0s + 1 * sW0, W0s + 2 * sW0,
        U0s + 0 * sU,  U0s + 1 * sU,
        bW0 + 0, bU0 + 0, bW0 + 1024, bU0 + 1024, bW0 + 2048,
        zws, gx2, rh);
    gru_cand<<<16, 128, 0, stream>>>(
        rh, U0s + 2 * sU, bU0 + 2048, gx2, zws, h0f, h0f, h0b, 0);
    // layer 1: x = new h0; NOTE reference swap: r uses Ul[1], z uses Ul[0]
    gru_gates<32><<<16, 128, 0, stream>>>(
        h0b, h1b, h1f,
        Wls + 0 * sU, Wls + 1 * sU, Wls + 2 * sU,
        Uls + 1 * sU /*Ur=Ul[1]*/, Uls + 0 * sU /*Uz=Ul[0]*/,
        bWl + 0, bUl + 1024, bWl + 1024, bUl + 0, bWl + 2048,
        zws, gx2, rh);
    gru_cand<<<16, 128, 0, stream>>>(
        rh, Uls + 2 * sU, bUl + 2048, gx2, zws, h1f, h1f, h1b, 1);
    // logits for this timestep
    gru_logits<<<157, 128, 0, stream>>>(h1b, Wys, by, out + (size_t)t * 64 * 10000);
  }

  // final hidden state appended after logits in d_out
  copy_h<<<512, 256, 0, stream>>>(hf, out + (size_t)128 * 64 * 10000, 2 * 64 * 1024);
}
